// QWenBlock_10814727651571
// MI455X (gfx1250) — matrix operand order repaired and checked by reading, not compiled, not run on hardware
//
#include <hip/hip_runtime.h>
#include <hip/hip_bf16.h>
#include <stdint.h>

// ---------------- problem constants ----------------
#define BQ   2
#define SQ   2048
#define HQ   2048
#define NHQ  16
#define HDQ  128
#define IQ   5504            // 11008/2
#define EPSQ 1e-6f
#define MQ   (BQ*SQ)         // 4096 token rows

typedef __bf16 bf16;
typedef __bf16 v16bf __attribute__((ext_vector_type(16)));
typedef float  v8f   __attribute__((ext_vector_type(8)));

union AFrag { v16bf v; uint32_t u[8]; bf16 h[16]; };

__device__ inline v8f vzero8() {
  v8f z;
#pragma unroll
  for (int i = 0; i < 8; ++i) z[i] = 0.0f;
  return z;
}

__device__ inline v8f wmma_bf16(const AFrag& a, const AFrag& b, v8f c) {
  // D = A*B + C, f32 accumulate, 16x16x32 bf16 (wave32)
  return __builtin_amdgcn_wmma_f32_16x16x32_bf16(false, a.v, false, b.v,
                                                 (short)0, c, false, false);
}

// ---------------- Tensor Data Mover (gfx1250) ----------------
#if defined(__AMDGCN__) && __has_builtin(__builtin_amdgcn_tensor_load_to_lds)
#define HAVE_TDM 1
typedef unsigned int u32x4 __attribute__((ext_vector_type(4)));
typedef int          i32x4 __attribute__((ext_vector_type(4)));
typedef int          i32x8 __attribute__((ext_vector_type(8)));

// 2-D tile DMA: global (row-major, 2-byte elems, row stride in elems) -> LDS
// (tile rows stored contiguously). D# packing per CDNA5 ISA ch.8 (§8.3/8.4).
__device__ inline void tdm_load_tile_bf16(void* lds_dst, const void* gsrc,
                                          uint32_t tile_w, uint32_t tile_h,
                                          uint64_t row_stride,
                                          uint64_t tensor_w, uint64_t tensor_h) {
  uint64_t ga = (uint64_t)(uintptr_t)gsrc;     // 57-bit global byte address
  uint32_t la = (uint32_t)(uintptr_t)lds_dst;  // LDS byte address (low 32 bits)
  u32x4 g0;
  g0[0] = 1u;                                           // count=1, user D#
  g0[1] = la;                                           // lds_addr
  g0[2] = (uint32_t)ga;                                 // global_addr[31:0]
  g0[3] = (uint32_t)((ga >> 32) & 0x01FFFFFFu) | (2u << 30);  // addr[56:32], type=2
  i32x8 g1;
  g1[0] = (int)(1u << 16);                              // data_size=1 (2B), mask=0
  g1[1] = (int)((uint32_t)(tensor_w & 0xFFFFu) << 16);  // tensor_dim0[15:0]
  g1[2] = (int)((uint32_t)((tensor_w >> 16) & 0xFFFFu) |
                ((uint32_t)(tensor_h & 0xFFFFu) << 16));        // dim0[31:16], dim1[15:0]
  g1[3] = (int)((uint32_t)((tensor_h >> 16) & 0xFFFFu) |
                (tile_w << 16));                                // dim1[31:16], tile_dim0
  g1[4] = (int)(tile_h & 0xFFFFu);                      // tile_dim1, tile_dim2=0
  g1[5] = (int)(uint32_t)(row_stride & 0xFFFFFFFFu);    // dim0_stride[31:0]
  g1[6] = (int)(uint32_t)((row_stride >> 32) & 0xFFFFu);// dim0_stride[47:32]
  g1[7] = 0;
  i32x4 z4 = {0, 0, 0, 0};
#if __clang_major__ >= 23
  i32x8 z8 = {0, 0, 0, 0, 0, 0, 0, 0};
  __builtin_amdgcn_tensor_load_to_lds(g0, g1, z4, z4, z8, 0);
#else
  __builtin_amdgcn_tensor_load_to_lds(g0, g1, z4, z4, 0);
#endif
}
__device__ inline void tdm_wait() { __builtin_amdgcn_s_wait_tensorcnt(0); }
#else
#define HAVE_TDM 0
#endif

// A fragment (16x32 bf16, row-major source, stride in elements).
// Lane L<16: row=row0+L, K = 0..15; lane L>=16: K = 16..31. Pairs are
// consecutive K, so each VGPR is one aligned dword from the row.
__device__ inline AFrag load_a_frag(const bf16* __restrict__ base, int stride,
                                    int row0, int k0, int lane) {
  AFrag f;
  int hi = lane >> 4;
  const bf16* p = base + (size_t)(row0 + (lane & 15)) * stride + k0 + 16 * hi;
  const uint32_t* q0 = (const uint32_t*)p;
  const uint32_t* q1 = (const uint32_t*)(p + 8);
  f.u[0] = q0[0]; f.u[1] = q0[1]; f.u[2] = q0[2]; f.u[3] = q0[3];
  f.u[4] = q1[0]; f.u[5] = q1[1]; f.u[6] = q1[2]; f.u[7] = q1[3];
  return f;
}

// B fragment (32x16 bf16) from a K-contiguous (column-major) buffer:
// base[(col)*stride + k].  Lane n<16: col=col0+n, K=0..15; lane>=16: K=16..31.
__device__ inline AFrag load_b_frag(const bf16* __restrict__ base, int stride,
                                    int col0, int k0, int lane) {
  AFrag f;
  int hi = lane >> 4;
  const uint32_t* q =
      (const uint32_t*)(base + (size_t)(col0 + (lane & 15)) * stride + k0 + 16 * hi);
#pragma unroll
  for (int j = 0; j < 8; ++j) f.u[j] = q[j];
  return f;
}

// ---------------- RMSNorm (fp32 in, bf16 out) ----------------
__global__ __launch_bounds__(256)
void rmsnorm_kernel(const float* __restrict__ x, const float* __restrict__ w,
                    bf16* __restrict__ out) {
  __shared__ float red[256];
  int row = blockIdx.x;
  const float* xr = x + (size_t)row * HQ;
  float s = 0.0f;
  for (int i = threadIdx.x; i < HQ; i += 256) { float v = xr[i]; s += v * v; }
  red[threadIdx.x] = s;
  __syncthreads();
  for (int off = 128; off > 0; off >>= 1) {
    if ((int)threadIdx.x < off) red[threadIdx.x] += red[threadIdx.x + off];
    __syncthreads();
  }
  float rs = rsqrtf(red[0] / (float)HQ + EPSQ);
  bf16* orow = out + (size_t)row * HQ;
  for (int i = threadIdx.x; i < HQ; i += 256)
    orow[i] = (bf16)(xr[i] * rs * w[i]);
}

// ---------------- GEMM: bf16 A (MxK) x fp32 W (KxN) -> epilogue -------------
// Block: 256 threads = 8 waves, tile M=128 N=64, K-step 32, double-buffered:
// TDM DMA of A tile i+1 and B-tile conversion overlap WMMA on tile i; one
// barrier per iteration.
// EPI 0: QKV split+bias -> q/k/v bf16 (B,NH,S,HD) layouts.
// EPI 1: outF = addend + acc (fp32).
template <int EPI>
__global__ __launch_bounds__(256)
void gemm_kernel(const bf16* __restrict__ A, const float* __restrict__ W,
                 const float* __restrict__ bias, const float* __restrict__ addend,
                 float* __restrict__ outF, bf16* __restrict__ qout,
                 bf16* __restrict__ kout, bf16* __restrict__ vout,
                 int K, int Nw) {
  __shared__ bf16 As[2][128 * 32];   // row-major [m][k]
  __shared__ bf16 Bs[2][64 * 32];    // column-major [n][k] (bf16-converted)

  int t = threadIdx.x, lane = t & 31, wave = t >> 5;
  int wm = wave >> 1, wn = wave & 1;        // 4 x 2 wave grid
  int m0 = blockIdx.y * 128;
  int n0 = blockIdx.x * 64;
  int nt = K / 32;

  v8f acc[2][2];
#pragma unroll
  for (int i = 0; i < 2; ++i)
#pragma unroll
    for (int j = 0; j < 2; ++j) acc[i][j] = vzero8();

  int arow = t >> 1, ahalf = (t & 1) * 16;  // A stager (fallback path)
  int bk = t >> 3, bn = (t & 7) * 8;        // B stager: 8 fp32 per thread
  (void)arow; (void)ahalf;

  auto stage_A = [&](int ti, bf16* dst) {
#if HAVE_TDM
    if (wave == 0)
      tdm_load_tile_bf16(dst, A + (size_t)m0 * K + ti * 32, /*tile_w=*/32,
                         /*tile_h=*/128, (uint64_t)K, (uint64_t)K, (uint64_t)MQ);
#else
    const uint4* ag = (const uint4*)(A + (size_t)(m0 + arow) * K + ti * 32 + ahalf);
    uint4 a0 = ag[0], a1 = ag[1];
    *((uint4*)(dst + arow * 32 + ahalf))     = a0;
    *((uint4*)(dst + arow * 32 + ahalf + 8)) = a1;
#endif
  };
  auto stage_B = [&](int ti, bf16* dst) {
    const float* wg = W + (size_t)(ti * 32 + bk) * Nw + n0 + bn;
    float4 w0 = ((const float4*)wg)[0];
    float4 w1 = ((const float4*)wg)[1];
    dst[(bn + 0) * 32 + bk] = (bf16)w0.x;
    dst[(bn + 1) * 32 + bk] = (bf16)w0.y;
    dst[(bn + 2) * 32 + bk] = (bf16)w0.z;
    dst[(bn + 3) * 32 + bk] = (bf16)w0.w;
    dst[(bn + 4) * 32 + bk] = (bf16)w1.x;
    dst[(bn + 5) * 32 + bk] = (bf16)w1.y;
    dst[(bn + 6) * 32 + bk] = (bf16)w1.z;
    dst[(bn + 7) * 32 + bk] = (bf16)w1.w;
  };

  // prologue: tile 0 in flight
  stage_A(0, As[0]);
  stage_B(0, Bs[0]);

  for (int ti = 0; ti < nt; ++ti) {
#if HAVE_TDM
    tdm_wait();                       // tile ti DMA complete (issuer wave)
#endif
    __syncthreads();                  // publish tile ti; retire reads of buf^1
    if (ti + 1 < nt) {                // overlap next tile with compute
      stage_A(ti + 1, As[(ti + 1) & 1]);
      stage_B(ti + 1, Bs[(ti + 1) & 1]);
      if (ti + 2 < nt)
        __builtin_prefetch(W + (size_t)((ti + 2) * 32 + bk) * Nw + n0 + bn, 0, 1);
    }
    const bf16* as = As[ti & 1];
    const bf16* bs = Bs[ti & 1];
    AFrag af[2], bfr[2];
#pragma unroll
    for (int i = 0; i < 2; ++i) af[i]  = load_a_frag(as, 32, wm * 32 + i * 16, 0, lane);
#pragma unroll
    for (int j = 0; j < 2; ++j) bfr[j] = load_b_frag(bs, 32, wn * 32 + j * 16, 0, lane);
#pragma unroll
    for (int i = 0; i < 2; ++i)
#pragma unroll
      for (int j = 0; j < 2; ++j) acc[i][j] = wmma_bf16(af[i], bfr[j], acc[i][j]);
  }

  int hi = lane >> 4, ln = lane & 15;
#pragma unroll
  for (int i = 0; i < 2; ++i) {
#pragma unroll
    for (int j = 0; j < 2; ++j) {
#pragma unroll
      for (int r = 0; r < 8; ++r) {
        int gm = m0 + wm * 32 + i * 16 + r + 8 * hi;
        int gn = n0 + wn * 32 + j * 16 + ln;
        float v = acc[i][j][r];
        if constexpr (EPI == 0) {
          v += bias[gn];
          int sel = gn / HQ;            // 0=q 1=k 2=v
          int rem = gn % HQ;
          int hh = rem / HDQ, d = rem % HDQ;
          int bb = gm / SQ, ss = gm % SQ;
          size_t dst = (((size_t)bb * NHQ + hh) * SQ + ss) * HDQ + d;
          bf16* dp = (sel == 0) ? qout : (sel == 1) ? kout : vout;
          dp[dst] = (bf16)v;
        } else {
          size_t o = (size_t)gm * HQ + gn;
          outF[o] = addend[o] + v;
        }
      }
    }
  }
}

// ---------------- gate_up GEMM with fused SiLU*up -> bf16 act ---------------
__global__ __launch_bounds__(256)
void gemm_gateup_kernel(const bf16* __restrict__ A, const float* __restrict__ W,
                        bf16* __restrict__ act, int K) {
  __shared__ bf16 As[2][128 * 32];
  __shared__ bf16 Bg[2][64 * 32];
  __shared__ bf16 Bu[2][64 * 32];

  int t = threadIdx.x, lane = t & 31, wave = t >> 5;
  int wm = wave >> 1, wn = wave & 1;
  int m0 = blockIdx.y * 128;
  int n0 = blockIdx.x * 64;               // gate column tile, up at +IQ
  const int Nw = 2 * IQ;
  int nt = K / 32;

  v8f accg[2][2], accu[2][2];
#pragma unroll
  for (int i = 0; i < 2; ++i)
#pragma unroll
    for (int j = 0; j < 2; ++j) { accg[i][j] = vzero8(); accu[i][j] = vzero8(); }

  int arow = t >> 1, ahalf = (t & 1) * 16;
  int bk = t >> 3, bn = (t & 7) * 8;
  (void)arow; (void)ahalf;

  auto stage_A = [&](int ti, bf16* dst) {
#if HAVE_TDM
    if (wave == 0)
      tdm_load_tile_bf16(dst, A + (size_t)m0 * K + ti * 32, 32, 128,
                         (uint64_t)K, (uint64_t)K, (uint64_t)MQ);
#else
    const uint4* ag = (const uint4*)(A + (size_t)(m0 + arow) * K + ti * 32 + ahalf);
    uint4 a0 = ag[0], a1 = ag[1];
    *((uint4*)(dst + arow * 32 + ahalf))     = a0;
    *((uint4*)(dst + arow * 32 + ahalf + 8)) = a1;
#endif
  };
  auto stage_GU = [&](int ti, bf16* dg, bf16* du) {
    const float* wg = W + (size_t)(ti * 32 + bk) * Nw + n0 + bn;
    const float* wu = wg + IQ;
    float4 g0 = ((const float4*)wg)[0];
    float4 g1 = ((const float4*)wg)[1];
    float4 u0 = ((const float4*)wu)[0];
    float4 u1 = ((const float4*)wu)[1];
    dg[(bn + 0) * 32 + bk] = (bf16)g0.x; dg[(bn + 1) * 32 + bk] = (bf16)g0.y;
    dg[(bn + 2) * 32 + bk] = (bf16)g0.z; dg[(bn + 3) * 32 + bk] = (bf16)g0.w;
    dg[(bn + 4) * 32 + bk] = (bf16)g1.x; dg[(bn + 5) * 32 + bk] = (bf16)g1.y;
    dg[(bn + 6) * 32 + bk] = (bf16)g1.z; dg[(bn + 7) * 32 + bk] = (bf16)g1.w;
    du[(bn + 0) * 32 + bk] = (bf16)u0.x; du[(bn + 1) * 32 + bk] = (bf16)u0.y;
    du[(bn + 2) * 32 + bk] = (bf16)u0.z; du[(bn + 3) * 32 + bk] = (bf16)u0.w;
    du[(bn + 4) * 32 + bk] = (bf16)u1.x; du[(bn + 5) * 32 + bk] = (bf16)u1.y;
    du[(bn + 6) * 32 + bk] = (bf16)u1.z; du[(bn + 7) * 32 + bk] = (bf16)u1.w;
  };

  stage_A(0, As[0]);
  stage_GU(0, Bg[0], Bu[0]);

  for (int ti = 0; ti < nt; ++ti) {
#if HAVE_TDM
    tdm_wait();
#endif
    __syncthreads();
    if (ti + 1 < nt) {
      stage_A(ti + 1, As[(ti + 1) & 1]);
      stage_GU(ti + 1, Bg[(ti + 1) & 1], Bu[(ti + 1) & 1]);
      if (ti + 2 < nt)
        __builtin_prefetch(W + (size_t)((ti + 2) * 32 + bk) * Nw + n0 + bn, 0, 1);
    }
    const bf16* as = As[ti & 1];
    AFrag af[2], bg[2], bu[2];
#pragma unroll
    for (int i = 0; i < 2; ++i) af[i] = load_a_frag(as, 32, wm * 32 + i * 16, 0, lane);
#pragma unroll
    for (int j = 0; j < 2; ++j) {
      bg[j] = load_b_frag(Bg[ti & 1], 32, wn * 32 + j * 16, 0, lane);
      bu[j] = load_b_frag(Bu[ti & 1], 32, wn * 32 + j * 16, 0, lane);
    }
#pragma unroll
    for (int i = 0; i < 2; ++i)
#pragma unroll
      for (int j = 0; j < 2; ++j) {
        accg[i][j] = wmma_bf16(af[i], bg[j], accg[i][j]);
        accu[i][j] = wmma_bf16(af[i], bu[j], accu[i][j]);
      }
  }

  int hi = lane >> 4, ln = lane & 15;
#pragma unroll
  for (int i = 0; i < 2; ++i)
#pragma unroll
    for (int j = 0; j < 2; ++j)
#pragma unroll
      for (int r = 0; r < 8; ++r) {
        int gm = m0 + wm * 32 + i * 16 + r + 8 * hi;
        int gn = n0 + wn * 32 + j * 16 + ln;
        float g = accg[i][j][r], u = accu[i][j][r];
        float sv = g / (1.0f + __expf(-g));        // SiLU
        act[(size_t)gm * IQ + gn] = (bf16)(sv * u);
      }
}

// ---------------- NeoX RoPE in place on q,k (bf16) ----------------
__global__ __launch_bounds__(64)
void rope_kernel(const int* __restrict__ pos, bf16* __restrict__ Q,
                 bf16* __restrict__ Kb) {
  int d = threadIdx.x;                 // 0..63 (pairs d, d+64)
  int s = blockIdx.x % SQ;
  int bh = blockIdx.x / SQ;            // b*NH + h
  size_t base = ((size_t)bh * SQ + s) * HDQ;
  float p = (float)pos[s];
  float inv = __expf(-((float)(2 * d) / (float)HDQ) * __logf(10000.0f));
  float fr = p * inv;
  float c = __cosf(fr), sn = __sinf(fr);
  {
    float x1 = (float)Q[base + d], x2 = (float)Q[base + d + 64];
    Q[base + d]      = (bf16)(x1 * c - x2 * sn);
    Q[base + d + 64] = (bf16)(x2 * c + x1 * sn);
  }
  {
    float x1 = (float)Kb[base + d], x2 = (float)Kb[base + d + 64];
    Kb[base + d]      = (bf16)(x1 * c - x2 * sn);
    Kb[base + d + 64] = (bf16)(x2 * c + x1 * sn);
  }
}

// ---------------- causal flash attention (bf16 in, bf16 out) ----------------
// Block: 128 threads = 4 waves; each wave owns 16 query rows; K/V tiles of 32
// keys double-buffered in LDS via TDM; P converted C-layout -> A-layout via
// per-wave LDS scratch.
__global__ __launch_bounds__(128)
void attn_kernel(const bf16* __restrict__ Qb, const bf16* __restrict__ Kb,
                 const bf16* __restrict__ Vb, bf16* __restrict__ AO) {
  __shared__ bf16 Ks[2][32 * HDQ];
  __shared__ bf16 Vs[2][32 * HDQ];
  __shared__ bf16 Pw[4 * 16 * 32];

  int t = threadIdx.x, lane = t & 31, wave = t >> 5;
  int qt = blockIdx.x & 31;            // S/64 = 32 query tiles
  int bh = blockIdx.x >> 5;            // b*NH + h
  int qbase = qt * 64;
  int qrow0 = qbase + wave * 16;

  const bf16* Qp = Qb + (size_t)bh * SQ * HDQ;
  const bf16* Kp = Kb + (size_t)bh * SQ * HDQ;
  const bf16* Vp = Vb + (size_t)bh * SQ * HDQ;

  AFrag qf[4];
#pragma unroll
  for (int c = 0; c < 4; ++c) qf[c] = load_a_frag(Qp, HDQ, qrow0, c * 32, lane);

  v8f o[8];
#pragma unroll
  for (int dt = 0; dt < 8; ++dt) o[dt] = vzero8();
  float mrow[8], lrow[8];
#pragma unroll
  for (int r = 0; r < 8; ++r) { mrow[r] = -1e30f; lrow[r] = 0.0f; }

  int hi = lane >> 4, ln = lane & 15;
  const float scale = 0.088388347648318447f;     // 1/sqrt(128)
  int srow = t >> 2, scol = (t & 3) * 32;        // fallback K/V stager
  (void)srow; (void)scol;
  int nkt = (qbase + 64) / 32;                   // causal tile count

  auto stage_KV = [&](int ti, bf16* kd, bf16* vd) {
#if HAVE_TDM
    if (wave == 0) {
      tdm_load_tile_bf16(kd, Kp + (size_t)ti * 32 * HDQ, HDQ, 32, HDQ,
                         (uint64_t)HDQ, (uint64_t)SQ);
      tdm_load_tile_bf16(vd, Vp + (size_t)ti * 32 * HDQ, HDQ, 32, HDQ,
                         (uint64_t)HDQ, (uint64_t)SQ);
    }
#else
    const uint4* kg = (const uint4*)(Kp + (size_t)(ti * 32 + srow) * HDQ + scol);
    const uint4* vg = (const uint4*)(Vp + (size_t)(ti * 32 + srow) * HDQ + scol);
    uint4* kdst = (uint4*)(kd + srow * HDQ + scol);
    uint4* vdst = (uint4*)(vd + srow * HDQ + scol);
#pragma unroll
    for (int j = 0; j < 4; ++j) kdst[j] = kg[j];
#pragma unroll
    for (int j = 0; j < 4; ++j) vdst[j] = vg[j];
#endif
  };

  stage_KV(0, Ks[0], Vs[0]);

  for (int ti = 0; ti < nkt; ++ti) {
    int kb = ti * 32;
#if HAVE_TDM
    tdm_wait();
#endif
    __syncthreads();
    if (ti + 1 < nkt)
      stage_KV(ti + 1, Ks[(ti + 1) & 1], Vs[(ti + 1) & 1]);
    const bf16* ks = Ks[ti & 1];
    const bf16* vs = Vs[ti & 1];

    // scores: two 16x16 tiles over this 32-key block (contract HD=128)
    v8f sc[2];
#pragma unroll
    for (int st = 0; st < 2; ++st) {
      sc[st] = vzero8();
#pragma unroll
      for (int c = 0; c < 4; ++c) {
        AFrag bfr = load_b_frag(ks, HDQ, st * 16, c * 32, lane);
        sc[st] = wmma_bf16(qf[c], bfr, sc[st]);
      }
    }

    // online softmax in C layout: row m = r + 8*hi, key n = lane&15
    bf16* pw = Pw + wave * 16 * 32;
#pragma unroll
    for (int r = 0; r < 8; ++r) {
      int q = qrow0 + r + 8 * hi;
      float s0 = ((kb + ln) <= q)      ? sc[0][r] * scale : -1e30f;
      float s1 = ((kb + 16 + ln) <= q) ? sc[1][r] * scale : -1e30f;
      float tmax = fmaxf(s0, s1);
#pragma unroll
      for (int off = 1; off < 16; off <<= 1)
        tmax = fmaxf(tmax, __shfl_xor(tmax, off, 32));
      float newm = fmaxf(mrow[r], tmax);
      float p0 = __expf(s0 - newm), p1 = __expf(s1 - newm);
      float ps = p0 + p1;
#pragma unroll
      for (int off = 1; off < 16; off <<= 1) ps += __shfl_xor(ps, off, 32);
      float fac = __expf(mrow[r] - newm);
      lrow[r] = lrow[r] * fac + ps;
      mrow[r] = newm;
#pragma unroll
      for (int dt = 0; dt < 8; ++dt) o[dt][r] *= fac;
      int m = r + 8 * hi;
      pw[m * 32 + ln]      = (bf16)p0;
      pw[m * 32 + 16 + ln] = (bf16)p1;
    }
    __syncthreads();                  // publish Pw (and order Vs reads)

    // P (16x32) @ V (32x128): 8 WMMAs over head-dim tiles
    AFrag pf = load_a_frag(Pw + wave * 16 * 32, 32, 0, 0, lane);
#pragma unroll
    for (int dt = 0; dt < 8; ++dt) {
      AFrag vf;
#pragma unroll
      for (int e = 0; e < 16; ++e)
        vf.h[e] = vs[(e + 16 * hi) * HDQ + dt * 16 + ln];
      o[dt] = wmma_bf16(pf, vf, o[dt]);
    }
  }

  // write attention output as (B,S,H) bf16 for the O-proj GEMM
  int b = bh / NHQ, h = bh % NHQ;
#pragma unroll
  for (int r = 0; r < 8; ++r) {
    float invl = 1.0f / lrow[r];
    int q = qrow0 + r + 8 * hi;
    size_t rowo = ((size_t)b * SQ + q) * HQ + (size_t)h * HDQ;
#pragma unroll
    for (int dt = 0; dt < 8; ++dt)
      AO[rowo + dt * 16 + ln] = (bf16)(o[dt][r] * invl);
  }
}

// ---------------- launch ----------------
extern "C" void kernel_launch(void* const* d_in, const int* in_sizes, int n_in,
                              void* d_out, int out_size, void* d_ws, size_t ws_size,
                              hipStream_t stream) {
  const int*   positions = (const int*)  d_in[0];
  const float* hidden    = (const float*)d_in[1];
  const float* ln1_w     = (const float*)d_in[2];
  const float* w_qkv     = (const float*)d_in[3];
  const float* b_qkv     = (const float*)d_in[4];
  const float* w_o       = (const float*)d_in[5];
  const float* ln2_w     = (const float*)d_in[6];
  const float* w_gate_up = (const float*)d_in[7];
  const float* w_down    = (const float*)d_in[8];
  float* out = (float*)d_out;

  char* ws = (char*)d_ws;
  size_t off = 0;
  auto alloc = [&](size_t bytes) {
    char* p = ws + off;
    off += (bytes + 255) & ~(size_t)255;
    return p;
  };
  bf16* X1  = (bf16*)alloc((size_t)MQ * HQ * 2);
  bf16* Qb  = (bf16*)alloc((size_t)MQ * HQ * 2);
  bf16* Kbf = (bf16*)alloc((size_t)MQ * HQ * 2);
  bf16* Vb  = (bf16*)alloc((size_t)MQ * HQ * 2);
  bf16* AO  = (bf16*)alloc((size_t)MQ * HQ * 2);
  float* H1 = (float*)alloc((size_t)MQ * HQ * 4);
  bf16* X2  = (bf16*)alloc((size_t)MQ * HQ * 2);
  bf16* ACT = (bf16*)alloc((size_t)MQ * IQ * 2);

  // 1) x = rmsnorm(hidden) -> bf16
  rmsnorm_kernel<<<MQ, 256, 0, stream>>>(hidden, ln1_w, X1);

  // 2) qkv = x @ w_qkv + b, scattered to q/k/v (B,NH,S,HD) bf16
  dim3 gq(3 * HQ / 64, MQ / 128);
  gemm_kernel<0><<<gq, 256, 0, stream>>>(X1, w_qkv, b_qkv, nullptr, nullptr,
                                         Qb, Kbf, Vb, HQ, 3 * HQ);

  // 3) RoPE (NeoX) in place on q,k
  rope_kernel<<<BQ * NHQ * SQ, 64, 0, stream>>>(positions, Qb, Kbf);

  // 4) causal flash attention -> AO (B,S,H) bf16
  attn_kernel<<<BQ * NHQ * (SQ / 64), 128, 0, stream>>>(Qb, Kbf, Vb, AO);

  // 5) h1 = hidden + AO @ w_o
  dim3 go(HQ / 64, MQ / 128);
  gemm_kernel<1><<<go, 256, 0, stream>>>(AO, w_o, nullptr, hidden, H1,
                                         nullptr, nullptr, nullptr, HQ, HQ);

  // 6) x2 = rmsnorm(h1) -> bf16
  rmsnorm_kernel<<<MQ, 256, 0, stream>>>(H1, ln2_w, X2);

  // 7) act = silu(x2@Wg) * (x2@Wu) -> bf16
  dim3 gg(IQ / 64, MQ / 128);
  gemm_gateup_kernel<<<gg, 256, 0, stream>>>(X2, w_gate_up, ACT, HQ);

  // 8) out = h1 + act @ w_down
  dim3 gd(HQ / 64, MQ / 128);
  gemm_kernel<1><<<gd, 256, 0, stream>>>(ACT, w_down, nullptr, H1, out,
                                         nullptr, nullptr, nullptr, IQ, HQ);
}
